// rnn_decoder_74294344286647
// MI455X (gfx1250) — matrix operand order repaired and checked by reading, not compiled, not run on hardware
//
#include <hip/hip_runtime.h>
#include <hip/hip_bf16.h>
#include <math.h>

// Sizes from the reference
#define B_ 32
#define T_ 64
#define S_ 64
#define E_ 512
#define H_ 1024

typedef __bf16 bf16;
typedef __attribute__((ext_vector_type(16))) __bf16 v16bf;
typedef __attribute__((ext_vector_type(8)))  __bf16 bf16x8;
typedef __attribute__((ext_vector_type(4)))  __bf16 bf16x4;
typedef __attribute__((ext_vector_type(8)))  float  v8f;

__device__ __forceinline__ float sigmoidf_(float x) { return 1.0f / (1.0f + expf(-x)); }

// A fragment (16-bit A 16x32 layout, half-lane offset folded into ptr):
// elems 0-7 = K[+0..7], elems 8-15 = K[+16..23]
__device__ __forceinline__ v16bf ldA(const bf16* __restrict__ p) {
  bf16x8 lo = *(const bf16x8*)(p);
  bf16x8 hi = *(const bf16x8*)(p + 16);
  return __builtin_shufflevector(lo, hi, 0,1,2,3,4,5,6,7,8,9,10,11,12,13,14,15);
}

// B fragment (32x16 layout, half-lane offset folded into ptr):
// elems 0-7 = K[+0..7], elems 8-15 = K[+16..23]
__device__ __forceinline__ v16bf ldB(const bf16* __restrict__ p) {
  bf16x8 lo = *(const bf16x8*)(p);
  bf16x8 hi = *(const bf16x8*)(p + 16);
  return __builtin_shufflevector(lo, hi, 0,1,2,3,4,5,6,7,8,9,10,11,12,13,14,15);
}

__device__ __forceinline__ bf16x4 cvt4(float4 v) {
  bf16x4 r;
  r[0] = (bf16)v.x; r[1] = (bf16)v.y; r[2] = (bf16)v.z; r[3] = (bf16)v.w;
  return r;
}

// ---------------------------------------------------------------------------
// Prep kernels
// ---------------------------------------------------------------------------

// Gather embeddings straight to bf16: Xb[b][t][:] = bf16(emb[inputs[b][t]][:])
__global__ void k_embed(const int* __restrict__ inp, const float* __restrict__ emb,
                        bf16* __restrict__ Xb) {
  int idx = blockIdx.x * blockDim.x + threadIdx.x;     // over B*T*(E/4)
  if (idx >= B_ * T_ * (E_ / 4)) return;
  int e4 = idx % (E_ / 4);
  int bt = idx / (E_ / 4);
  int tok = inp[bt];
  float4 v = ((const float4*)emb)[(size_t)tok * (E_ / 4) + e4];
  ((bf16x4*)Xb)[idx] = cvt4(v);
}

// Concat-convert weights: dst[n][0:ka) = bf16(A[n]), dst[n][ka:ka+kb) = bf16(Bsrc[n])
__global__ void k_cat2(const float* __restrict__ Asrc, int ka,
                       const float* __restrict__ Bsrc, int kb,
                       bf16* __restrict__ dst, int nrows) {
  int k = ka + kb;
  int idx = blockIdx.x * blockDim.x + threadIdx.x;
  if (idx >= nrows * k) return;
  int col = idx % k;
  int row = idx / k;
  float v = (col < ka) ? Asrc[(size_t)row * ka + col]
                       : Bsrc[(size_t)row * kb + (col - ka)];
  dst[idx] = (bf16)v;
}

// b0[n] = x[n] + y[n]  (fold b_ih + b_hh)
__global__ void k_addb(const float* __restrict__ x, const float* __restrict__ y,
                       float* __restrict__ dst, int n) {
  int idx = blockIdx.x * blockDim.x + threadIdx.x;
  if (idx < n) dst[idx] = x[idx] + y[idx];
}

// ---------------------------------------------------------------------------
// LSTM cell elementwise (PyTorch gate order i,f,g,o).
// Updates f32 state and writes a bf16 mirror of h for the WMMA consumers.
// ---------------------------------------------------------------------------
__global__ void k_lstm(const float* __restrict__ gates,
                       float* __restrict__ h, float* __restrict__ c,
                       bf16* __restrict__ hB) {
  int idx = blockIdx.x * blockDim.x + threadIdx.x;
  if (idx >= B_ * H_) return;
  int hh = idx % H_;
  int b  = idx / H_;
  const float* g = gates + (size_t)b * 4 * H_;
  float ig = sigmoidf_(g[hh]);
  float fg = sigmoidf_(g[H_ + hh]);
  float gg = tanhf(g[2 * H_ + hh]);
  float og = sigmoidf_(g[3 * H_ + hh]);
  float c2 = fg * c[idx] + ig * gg;
  float h2 = og * tanhf(c2);
  c[idx] = c2;
  h[idx] = h2;
  hB[idx] = (bf16)h2;
}

// ---------------------------------------------------------------------------
// WMMA bf16 GEMM, 2Mx2N register blocking, all-bf16 operand loads.
//   out[m][n] = act( sum_k A[m][k] * W[n][k] + bias[n] )
// A is the logical concat [src1 (cols 0..ka) | src2 (cols ka..K)], both bf16.
// K-loop split at ka so each loop is branch-free with hoisted bases.
// W: [N][ldw] bf16. One wave per 32x32 output block. blockDim = 64 (2 waves).
// K and ka must be multiples of 32.
// ---------------------------------------------------------------------------
__global__ void k_gemm(const bf16* __restrict__ src1, int lds1, int ka,
                       const bf16* __restrict__ src2, int lds2,
                       const bf16* __restrict__ W, int ldw,
                       const float* __restrict__ bias,
                       float* __restrict__ out, int ldo, int outOff,
                       int K, int act) {
  int lane = threadIdx.x & 31;
  int wid  = (blockIdx.x * blockDim.x + threadIdx.x) >> 5;   // global wave id
  int nT0  = wid << 1;                                       // first of 2 N-tiles
  int ln = lane & 15;
  int hi = lane >> 4;
  int aoff = hi << 3;                                        // A half-lane K offset
  int m0 = ln;
  int m1 = 16 + ln;

  const bf16* Wr0 = W + (size_t)((nT0 << 4) + ln) * ldw + (hi << 3);
  const bf16* Wr1 = Wr0 + (size_t)16 * ldw;

  v8f acc00 = {}, acc01 = {}, acc10 = {}, acc11 = {};

  // ---- segment 1: A columns [0, ka) from src1 ----
  {
    const bf16* A0p = src1 + (size_t)m0 * lds1 + aoff;
    const bf16* A1p = src1 + (size_t)m1 * lds1 + aoff;
    for (int k0 = 0; k0 < ka; k0 += 32) {
      v16bf a0 = ldA(A0p + k0);
      v16bf a1 = ldA(A1p + k0);
      v16bf w0 = ldB(Wr0 + k0);
      v16bf w1 = ldB(Wr1 + k0);
      acc00 = __builtin_amdgcn_wmma_f32_16x16x32_bf16(false, a0, false, w0, (short)0, acc00, false, false);
      acc01 = __builtin_amdgcn_wmma_f32_16x16x32_bf16(false, a0, false, w1, (short)0, acc01, false, false);
      acc10 = __builtin_amdgcn_wmma_f32_16x16x32_bf16(false, a1, false, w0, (short)0, acc10, false, false);
      acc11 = __builtin_amdgcn_wmma_f32_16x16x32_bf16(false, a1, false, w1, (short)0, acc11, false, false);
    }
  }
  // ---- segment 2: A columns [ka, K) from src2 ----
  {
    const bf16* A0p = src2 + (size_t)m0 * lds2 + aoff;
    const bf16* A1p = src2 + (size_t)m1 * lds2 + aoff;
    const bf16* W0p = Wr0 + ka;
    const bf16* W1p = Wr1 + ka;
    int kb = K - ka;
    for (int k0 = 0; k0 < kb; k0 += 32) {
      v16bf a0 = ldA(A0p + k0);
      v16bf a1 = ldA(A1p + k0);
      v16bf w0 = ldB(W0p + k0);
      v16bf w1 = ldB(W1p + k0);
      acc00 = __builtin_amdgcn_wmma_f32_16x16x32_bf16(false, a0, false, w0, (short)0, acc00, false, false);
      acc01 = __builtin_amdgcn_wmma_f32_16x16x32_bf16(false, a0, false, w1, (short)0, acc01, false, false);
      acc10 = __builtin_amdgcn_wmma_f32_16x16x32_bf16(false, a1, false, w0, (short)0, acc10, false, false);
      acc11 = __builtin_amdgcn_wmma_f32_16x16x32_bf16(false, a1, false, w1, (short)0, acc11, false, false);
    }
  }

  // Epilogue: C/D layout -> VGPR r holds M = mBase + r, N = ln (per lane half)
  int n0 = (nT0 << 4) + ln;
  int n1 = n0 + 16;
  float bv0 = bias[n0];
  float bv1 = bias[n1];
  int mA = hi << 3;
  int mB = 16 + (hi << 3);
#pragma unroll
  for (int r = 0; r < 8; ++r) {
    float v00 = acc00[r] + bv0;
    float v01 = acc01[r] + bv1;
    float v10 = acc10[r] + bv0;
    float v11 = acc11[r] + bv1;
    if (act) { v00 = tanhf(v00); v01 = tanhf(v01); v10 = tanhf(v10); v11 = tanhf(v11); }
    out[(size_t)(mA + r) * ldo + outOff + n0] = v00;
    out[(size_t)(mA + r) * ldo + outOff + n1] = v01;
    out[(size_t)(mB + r) * ldo + outOff + n0] = v10;
    out[(size_t)(mB + r) * ldo + outOff + n1] = v11;
  }
}

// ---------------------------------------------------------------------------
// Luong attention: scores -> softmax -> bf16 context. One block per batch row.
// ---------------------------------------------------------------------------
__global__ void k_attn(const float* __restrict__ enc, const float* __restrict__ gamma,
                       bf16* __restrict__ ctxB) {
  __shared__ float red[256];
  __shared__ float sc[S_];
  int b = blockIdx.x;
  const float* encB = enc + (size_t)b * S_ * H_;
  const float* gam  = gamma + (size_t)b * H_;

  // scores: 4 threads per s, each sums a contiguous quarter of H (float4 loads)
  int s    = threadIdx.x >> 2;
  int part = threadIdx.x & 3;
  float acc = 0.0f;
  int hBeg = part * (H_ / 4);
  for (int h = hBeg; h < hBeg + H_ / 4; h += 4) {
    float4 e = *(const float4*)(encB + (size_t)s * H_ + h);
    float4 g = *(const float4*)(gam + h);
    acc += e.x * g.x + e.y * g.y + e.z * g.z + e.w * g.w;
  }
  red[threadIdx.x] = acc;
  __syncthreads();
  if (part == 0) sc[s] = red[s * 4] + red[s * 4 + 1] + red[s * 4 + 2] + red[s * 4 + 3];
  __syncthreads();

  // softmax over 64 scores (tiny; serialized in one lane)
  if (threadIdx.x == 0) {
    float m = -1e30f;
    for (int i = 0; i < S_; ++i) m = fmaxf(m, sc[i]);
    float sum = 0.0f;
    for (int i = 0; i < S_; ++i) { float e = expf(sc[i] - m); sc[i] = e; sum += e; }
    float inv = 1.0f / sum;
    for (int i = 0; i < S_; ++i) sc[i] *= inv;
  }
  __syncthreads();

  // context: thread i owns float4 at h = 4*i; emit bf16 mirror for the WMMA GEMM
  int h = threadIdx.x * 4;
  float4 cv = make_float4(0.f, 0.f, 0.f, 0.f);
  for (int s2 = 0; s2 < S_; ++s2) {
    float w = sc[s2];
    float4 e = *(const float4*)(encB + (size_t)s2 * H_ + h);
    cv.x += w * e.x; cv.y += w * e.y; cv.z += w * e.z; cv.w += w * e.w;
  }
  *(bf16x4*)(ctxB + (size_t)b * H_ + h) = cvt4(cv);
}

// ---------------------------------------------------------------------------
// Host launcher
// ---------------------------------------------------------------------------
extern "C" void kernel_launch(void* const* d_in, const int* in_sizes, int n_in,
                              void* d_out, int out_size, void* d_ws, size_t ws_size,
                              hipStream_t stream) {
  (void)in_sizes; (void)n_in; (void)out_size; (void)ws_size;

  const int*   inp  = (const int*)d_in[0];
  const float* enc  = (const float*)d_in[1];
  const float* h0   = (const float*)d_in[2];
  const float* c0   = (const float*)d_in[3];
  const float* emb  = (const float*)d_in[4];
  const float* Wih0 = (const float*)d_in[5];
  const float* bih0 = (const float*)d_in[6];
  const float* Whh0 = (const float*)d_in[7];
  const float* bhh0 = (const float*)d_in[8];
  const float* Wih1 = (const float*)d_in[9];
  const float* bih1 = (const float*)d_in[10];
  const float* Whh1 = (const float*)d_in[11];
  const float* bhh1 = (const float*)d_in[12];
  const float* Win  = (const float*)d_in[13];
  const float* bin  = (const float*)d_in[14];
  const float* Wout = (const float*)d_in[15];
  const float* bout = (const float*)d_in[16];
  float* out = (float*)d_out;

  // ---- workspace carve-up (~40 MB) ----
  char*  base = (char*)d_ws;
  size_t cur  = 0;
  auto alloc = [&](size_t bytes) -> char* {
    char* p = base + cur;
    cur = (cur + bytes + 255) & ~(size_t)255;
    return p;
  };
  bf16*  Xb    = (bf16*)alloc((size_t)B_ * T_ * E_ * 2);
  bf16*  W0    = (bf16*)alloc((size_t)4096 * 1536 * 2);
  bf16*  W1    = (bf16*)alloc((size_t)4096 * 2048 * 2);
  bf16*  WinB  = (bf16*)alloc((size_t)1024 * 1024 * 2);
  bf16*  WoutB = (bf16*)alloc((size_t)1024 * 2048 * 2);
  float* b0    = (float*)alloc(4096 * 4);
  float* b1    = (float*)alloc(4096 * 4);
  float* hS    = (float*)alloc((size_t)2 * B_ * H_ * 4);
  float* cS    = (float*)alloc((size_t)2 * B_ * H_ * 4);
  bf16*  hM    = (bf16*)alloc((size_t)2 * B_ * H_ * 2);   // bf16 mirrors of h
  float* gates = (float*)alloc((size_t)B_ * 4096 * 4);
  float* gamma = (float*)alloc((size_t)B_ * H_ * 4);
  bf16*  ctxB  = (bf16*)alloc((size_t)B_ * H_ * 2);

  float* h0S = hS;
  float* h1S = hS + B_ * H_;
  float* c0S = cS;
  float* c1S = cS + B_ * H_;
  bf16*  h0M = hM;
  bf16*  h1M = hM + B_ * H_;

  // ---- one-time prep (runs every call; deterministic) ----
  k_embed<<<(B_ * T_ * (E_ / 4) + 255) / 256, 256, 0, stream>>>(inp, emb, Xb);
  k_cat2<<<(4096 * 1536 + 255) / 256, 256, 0, stream>>>(Wih0, E_, Whh0, H_, W0, 4096);
  k_cat2<<<(4096 * 2048 + 255) / 256, 256, 0, stream>>>(Wih1, H_, Whh1, H_, W1, 4096);
  k_cat2<<<(1024 * 1024 + 255) / 256, 256, 0, stream>>>(Win, H_, nullptr, 0, WinB, 1024);
  k_cat2<<<(1024 * 2048 + 255) / 256, 256, 0, stream>>>(Wout, 2 * H_, nullptr, 0, WoutB, 1024);
  k_addb<<<(4096 + 255) / 256, 256, 0, stream>>>(bih0, bhh0, b0, 4096);
  k_addb<<<(4096 + 255) / 256, 256, 0, stream>>>(bih1, bhh1, b1, 4096);
  // initial states: f32 copy + bf16 mirror (reuse k_cat2 for the mirror convert)
  hipMemcpyAsync(hS, h0, (size_t)2 * B_ * H_ * 4, hipMemcpyDeviceToDevice, stream);
  hipMemcpyAsync(cS, c0, (size_t)2 * B_ * H_ * 4, hipMemcpyDeviceToDevice, stream);
  k_cat2<<<(2 * B_ * H_ + 255) / 256, 256, 0, stream>>>(h0, 2 * B_ * H_, nullptr, 0, hM, 1);

  const int eltGrid = (B_ * H_ + 255) / 256;

  // ---- sequential timestep chain (7 kernels/step) ----
  for (int t = 0; t < T_; ++t) {
    const bf16* Xt = Xb + (size_t)t * E_;   // row b at Xt + b*(T_*E_)

    // layer 0 gates: A = [x_t | h0], K = 1536, N = 4096
    k_gemm<<<4096 / 64, 64, 0, stream>>>(Xt, T_ * E_, E_, h0M, H_,
                                         W0, 1536, b0, gates, 4096, 0, 1536, 0);
    k_lstm<<<eltGrid, 256, 0, stream>>>(gates, h0S, c0S, h0M);

    // layer 1 gates: A = [h0' | h1_old] (stream order keeps h1 unclobbered)
    k_gemm<<<4096 / 64, 64, 0, stream>>>(h0M, H_, H_, h1M, H_,
                                         W1, 2048, b1, gates, 4096, 0, 2048, 0);
    k_lstm<<<eltGrid, 256, 0, stream>>>(gates, h1S, c1S, h1M);

    // gamma = h1 @ W_in^T + b_in, K = N = 1024 (segment 2 empty: ka == K)
    k_gemm<<<1024 / 64, 64, 0, stream>>>(h1M, H_, H_, h1M, H_,
                                         WinB, 1024, bin, gamma, 1024, 0, 1024, 0);

    // attention -> bf16 context
    k_attn<<<B_, 256, 0, stream>>>(enc, gamma, ctxB);

    // out_t = tanh([c_t | h1] @ W_out^T + b_out) -> hidden_outputs[:, t, :]
    k_gemm<<<1024 / 64, 64, 0, stream>>>(ctxB, H_, H_, h1M, H_,
                                         WoutB, 2048, bout, out, T_ * H_, t * H_, 2048, 1);
  }

  // final states: hT then cT appended after hidden_outputs
  float* outHT = out + (size_t)B_ * T_ * H_;
  float* outCT = outHT + (size_t)2 * B_ * H_;
  hipMemcpyAsync(outHT, hS, (size_t)2 * B_ * H_ * 4, hipMemcpyDeviceToDevice, stream);
  hipMemcpyAsync(outCT, cS, (size_t)2 * B_ * H_ * 4, hipMemcpyDeviceToDevice, stream);
}
